// HausdorffDistanceLoss_19310172963001
// MI455X (gfx1250) — compile-verified, matrix operand order checked by reading, not run on hardware
//
#include <hip/hip_runtime.h>

// ---------------------------------------------------------------------------
// Hausdorff distance loss for MI455X (gfx1250, wave32).
//
//   k_zero     : zero the 2 global accumulators in ws
//   k_rowpass  : argmax/binarize + exact 1D EDT pass along W -> g2 (squared
//                1D distance) + packed mask bits, one thread per row
//   k_colpass  : exact column pass dt2[y,x] = min_y' g2[y',x] + (y-y')^2 with
//                DOUBLE-BUFFERED async global->LDS tile staging
//                (global_load_async_to_lds_b128 + s_wait_asynccnt), then
//                masked-sum reduction via v_wmma_f32_16x16x32_f16
//                (hi/lo f16 split of dist, f32 accumulator)
//   k_final    : scalar combine
//
// ws layout: [0,8)   float sums[2]   (sums[0]=p2t numerator, sums[1]=t2p)
//            [256, 256+8MB)          g2   [32][256][256] f32
//            [256+8MB, +256KB)       mask bits [32][256][8] u32
// ---------------------------------------------------------------------------

typedef _Float16 v16h __attribute__((ext_vector_type(16)));
typedef float    v8f  __attribute__((ext_vector_type(8)));

// The async builtin takes pointers to 16-byte int vectors (per hipcc
// diagnostic): global (AS1) source, LDS (AS3) destination.
typedef int v4i_vs __attribute__((vector_size(16)));
typedef __attribute__((address_space(1))) v4i_vs gv4i;
typedef __attribute__((address_space(3))) v4i_vs lv4i;

#define NEG_SENTINEL (-1.0e6f)

// ---- gfx1250 async global->LDS helpers ------------------------------------
__device__ __forceinline__ void async_copy16(const void* g, void* l) {
#if __has_builtin(__builtin_amdgcn_global_load_async_to_lds_b128)
    __builtin_amdgcn_global_load_async_to_lds_b128(
        (gv4i*)(unsigned long long)(uintptr_t)g,
        (lv4i*)(unsigned int)(uintptr_t)l, 0, 0);
#else
    // GV mode: per-lane LDS byte offset in vdst, per-lane 64-bit global vaddr
    unsigned int       loff = (unsigned int)(uintptr_t)l;   // low 32 bits == LDS offset
    unsigned long long ga   = (unsigned long long)(uintptr_t)g;
    asm volatile("global_load_async_to_lds_b128 %0, %1, off"
                 :: "v"(loff), "v"(ga) : "memory");
#endif
}

__device__ __forceinline__ void wait_async_le8() {
#if __has_builtin(__builtin_amdgcn_s_wait_asynccnt)
    __builtin_amdgcn_s_wait_asynccnt(8);
#else
    asm volatile("s_wait_asynccnt 0x8" ::: "memory");
#endif
}

__device__ __forceinline__ void wait_async_le0() {
#if __has_builtin(__builtin_amdgcn_s_wait_asynccnt)
    __builtin_amdgcn_s_wait_asynccnt(0);
#else
    asm volatile("s_wait_asynccnt 0x0" ::: "memory");
#endif
}

__global__ __launch_bounds__(32) void k_zero(float* sums) {
    if (threadIdx.x < 2) sums[threadIdx.x] = 0.0f;
}

// One block per image-mask (b<16: pred argmax mask, b>=16: target mask),
// one thread per row. Forward scan writes dl, backward scan finishes g2.
__global__ __launch_bounds__(256) void k_rowpass(const float* __restrict__ pred,
                                                 const int*   __restrict__ tgt,
                                                 float*       __restrict__ g2,
                                                 unsigned*    __restrict__ mbits) {
    const int b = blockIdx.x;    // 0..31
    const int y = threadIdx.x;   // 0..255

    unsigned bits[8];
#pragma unroll
    for (int w = 0; w < 8; ++w) bits[w] = 0u;

    float* row = g2 + ((size_t)b * 256 + y) * 256;

    // forward: dl = x - (nearest fg index <= x, else -1e6)
    if (b < 16) {
        const float* p0 = pred + ((size_t)(b * 2 + 0) * 256 + y) * 256;
        const float* p1 = pred + ((size_t)(b * 2 + 1) * 256 + y) * 256;
        float last = NEG_SENTINEL;
        for (int x = 0; x < 256; ++x) {
            const bool m = p1[x] > p0[x];   // argmax==1 (ties -> 0, matches jnp.argmax)
            const float xf = (float)x;
            if (m) { last = xf; bits[x >> 5] |= (1u << (x & 31)); }
            row[x] = xf - last;
        }
    } else {
        const int* t = tgt + ((size_t)(b - 16) * 256 + y) * 256;
        float last = NEG_SENTINEL;
        for (int x = 0; x < 256; ++x) {
            const bool m = (t[x] == 1);
            const float xf = (float)x;
            if (m) { last = xf; bits[x >> 5] |= (1u << (x & 31)); }
            row[x] = xf - last;
        }
    }

    // backward: dr = (nearest fg index >= x, else +1e6) - x; g2 = min(dl,dr)^2
    float nextp = 1.0e6f;
    for (int x = 255; x >= 0; --x) {
        if ((bits[x >> 5] >> (x & 31)) & 1u) nextp = (float)x;
        const float dr = nextp - (float)x;
        const float d  = fminf(row[x], dr);
        row[x] = d * d;
    }

    unsigned* mb = mbits + ((size_t)b * 256 + y) * 8;
#pragma unroll
    for (int w = 0; w < 8; ++w) mb[w] = bits[w];
}

// One block per (image-mask, y-tile of 32). Thread = one column x with 32
// register accumulators. g2 rows double-buffered into LDS via async copies.
__global__ __launch_bounds__(256) void k_colpass(const float*    __restrict__ g2,
                                                 const unsigned* __restrict__ mbits,
                                                 float*          __restrict__ sums) {
    __shared__ __align__(16) float smem[16384];   // 64 KB: 2 x 32 KB tile buffers

    const int im    = blockIdx.x >> 3;   // 0..31
    const int ytile = blockIdx.x & 7;    // 0..7
    const int x     = threadIdx.x;       // 0..255
    const int ybase = ytile * 32;

    float acc[32];
#pragma unroll
    for (int i = 0; i < 32; ++i) acc[i] = 3.0e30f;

    const float* img = g2 + (size_t)im * 65536;

    // prologue: async-fill buffer 0 with chunk 0 (8 b128 issues per wave)
    {
        const float4* src = (const float4*)img;
        float4*       dst = (float4*)smem;
#pragma unroll
        for (int k = 0; k < 8; ++k) async_copy16(src + k * 256 + x, dst + k * 256 + x);
    }

    for (int c = 0; c < 8; ++c) {
        const float* tile = (c & 1) ? (smem + 8192) : smem;
        if (c < 7) {
            // prefetch chunk c+1 into the other buffer (safe: barrier at end of
            // previous iteration guarantees everyone finished reading it)
            const float4* src = (const float4*)(img + (size_t)(c + 1) * 8192);
            float4*       dst = (float4*)(((c + 1) & 1) ? (smem + 8192) : smem);
#pragma unroll
            for (int k = 0; k < 8; ++k) async_copy16(src + k * 256 + x, dst + k * 256 + x);
            wait_async_le8();   // chunk c complete; chunk c+1 may stay in flight
        } else {
            wait_async_le0();   // drain
        }
        __syncthreads();        // chunk c visible to all waves

#pragma unroll 4
        for (int yy = 0; yy < 32; ++yy) {
            const float g  = tile[yy * 256 + x];   // conflict-free column read
            const float yp = (float)(c * 32 + yy);
#pragma unroll
            for (int i = 0; i < 32; ++i) {
                const float d = (float)(ybase + i) - yp;
                acc[i] = fminf(acc[i], fmaf(d, d, g));
            }
        }
        __syncthreads();        // everyone done reading tile before it is refilled
    }

    // ---- masked-sum reduction via WMMA ------------------------------------
    // Multiply dt (sqrt of acc) by the OTHER source's mask at the same pixel.
    const int omImg = (im < 16) ? (im + 16) : (im - 16);
    const unsigned* mb = mbits + (size_t)omImg * 2048 + (x >> 5);

    const int lane = x & 31;
    const int wv   = x >> 5;              // 8 waves per block

    _Float16* base = (_Float16*)smem + wv * 3072;  // [dist_hi|dist_lo|mask] x 1024

#pragma unroll
    for (int i = 0; i < 32; ++i) {
        const int y = ybase + i;
        float dist = sqrtf(acc[i]);
        dist = fminf(dist, 60000.0f);               // keep hi part finite in f16
        const _Float16 h  = (_Float16)dist;
        const _Float16 lo = (_Float16)(dist - (float)h);   // exact residual
        const unsigned word = mb[(size_t)y * 8];
        const float mk = (float)((word >> (x & 31)) & 1u);
        const int p = lane * 32 + i;
        base[p]        = h;
        base[1024 + p] = lo;
        base[2048 + p] = (_Float16)mk;
    }
    __syncthreads();

    // Per wave: 1024 products as 2 x (16 rows x K=32) WMMAs, dots on diagonal.
    // A[m,k] = dist[seg*512 + m*32 + k], B[k,n] = mask[seg*512 + n*32 + k].
    v8f cacc = {};
    const int r   = lane & 15;
    const int hi8 = (lane >> 4) << 3;     // K offset 0 / 8 per documented layout
#pragma unroll
    for (int seg = 0; seg < 2; ++seg) {
        const _Float16* pD = base +        seg * 512;
        const _Float16* pL = base + 1024 + seg * 512;
        const _Float16* pM = base + 2048 + seg * 512;
        v16h a, al, bm;
#pragma unroll
        for (int h = 0; h < 16; ++h) {
            const int K   = ((h < 8) ? h : (h + 8)) + hi8;  // 16-bit A layout
            const int idx = r * 32 + K;
            a[h]  = pD[idx];
            al[h] = pL[idx];
            bm[h] = pM[idx];
        }
        cacc = __builtin_amdgcn_wmma_f32_16x16x32_f16(false, a,  false, bm,
                                                      (short)0, cacc, false, false);
        cacc = __builtin_amdgcn_wmma_f32_16x16x32_f16(false, al, false, bm,
                                                      (short)0, cacc, false, false);
    }

    // diagonal: D[q,q] in lane q (c[q]), D[q+8,q+8] in lane 24+q (c[q])
    float v = 0.0f;
#pragma unroll
    for (int q = 0; q < 8; ++q) {
        const float cq = cacc[q];
        v = (lane == q || lane == 24 + q) ? cq : v;
    }
#pragma unroll
    for (int off = 16; off > 0; off >>= 1) v += __shfl_xor(v, off, 32);

    if (lane == 0) atomicAdd(&sums[(im < 16) ? 1 : 0], v);
}

__global__ void k_final(const float* __restrict__ sums, float* __restrict__ out) {
    // loss = (S_p2t/N + S_t2p/N) / 2, N = 16*256*256, ALPHA = 1
    out[0] = (sums[0] + sums[1]) * (1.0f / (2.0f * 16.0f * 256.0f * 256.0f));
}

extern "C" void kernel_launch(void* const* d_in, const int* in_sizes, int n_in,
                              void* d_out, int out_size, void* d_ws, size_t ws_size,
                              hipStream_t stream) {
    const float* pred = (const float*)d_in[0];   // [16,2,256,256] f32
    const int*   tgt  = (const int*)d_in[1];     // [16,256,256]   i32

    float*    sums  = (float*)d_ws;
    float*    g2    = (float*)((char*)d_ws + 256);
    unsigned* mbits = (unsigned*)((char*)d_ws + 256 + (size_t)32 * 65536 * 4);

    k_zero   <<<1,   32,  0, stream>>>(sums);
    k_rowpass<<<32,  256, 0, stream>>>(pred, tgt, g2, mbits);
    k_colpass<<<256, 256, 0, stream>>>(g2, mbits, sums);
    k_final  <<<1,   1,   0, stream>>>(sums, (float*)d_out);
}
